// RnnSearchModule_76407468196169
// MI455X (gfx1250) — matrix operand order repaired and checked by reading, not compiled, not run on hardware
//
#include <hip/hip_runtime.h>

// Problem constants (from reference)
#define T_STEPS 512
#define BSZ     64
#define CIN     512
#define CH      1024

typedef __attribute__((ext_vector_type(16))) __bf16 v16bf;
typedef __attribute__((ext_vector_type(8)))  float  v8f;

union Frag { unsigned u[8]; v16bf v; };

#if defined(__AMDGCN__) && __has_builtin(__builtin_amdgcn_global_load_async_to_lds_b128)
#define HAVE_ASYNC_LDS 1
// Builtin signature (from hipcc diagnostic): (int __vector(4) __device__*,
// int __vector(4) __shared__*, Imm int offset, Imm int cpol).
typedef int v4i_gcc __attribute__((vector_size(16)));
typedef v4i_gcc __attribute__((address_space(1)))* glb_v4i_p;
typedef v4i_gcc __attribute__((address_space(3)))* lds_v4i_p;
#endif

__device__ __forceinline__ unsigned short f2bf(float f) {
  unsigned u = __float_as_uint(f);
  u += 0x7FFFu + ((u >> 16) & 1u);   // round-to-nearest-even
  return (unsigned short)(u >> 16);
}
__device__ __forceinline__ unsigned pack2(float lo, float hi) {
  return (unsigned)f2bf(lo) | ((unsigned)f2bf(hi) << 16);
}

// ---------------------------------------------------------------------------
// Prep: swizzle Wh (f32 [CH][CH]) into per-lane B-fragment layout (bf16 pairs).
// whSwz[((nt*32 + kc)*32 + lane)*8 + j]
//   lane: N = lane&15, Kbase = (lane<16)?0:8 ; u32 j holds K pair (Kbase+2j+(j>=4?8:0),+1)
// ---------------------------------------------------------------------------
__global__ void prep_wh_kernel(const float* __restrict__ Wh, unsigned* __restrict__ whSwz) {
  int idx = blockIdx.x * 256 + threadIdx.x;              // [0, 64*32*32*8)
  int j  = idx & 7;
  int l  = (idx >> 3) & 31;
  int kc = (idx >> 8) & 31;
  int nt = idx >> 13;
  int n  = nt * 16 + (l & 15);
  int k0 = kc * 32 + ((l < 16) ? 0 : 8) + 2 * j + ((j >= 4) ? 8 : 0);
  whSwz[idx] = pack2(Wh[(size_t)k0 * CH + n], Wh[(size_t)(k0 + 1) * CH + n]);
}

// ---------------------------------------------------------------------------
// Prep: h0 (f32 [BSZ][CH]) -> A-fragment layout into comm buffer 0.
// comm[((g*32 + kc)*32 + lane)*8 + j]
//   lane: M = lane&15, ks = (lane<16)?0:8
//   j<4: K pair (kc*32+ks+2j,+1) ; j>=4: K pair (kc*32+ks+16+2(j-4),+1)
// ---------------------------------------------------------------------------
__global__ void prep_h0_kernel(const float* __restrict__ h0, unsigned* __restrict__ comm) {
  int idx = blockIdx.x * 256 + threadIdx.x;              // [0, 4*32*32*8)
  int j  = idx & 7;
  int l  = (idx >> 3) & 31;
  int kc = (idx >> 8) & 31;
  int g  = idx >> 13;
  int m  = g * 16 + (l & 15);
  int ks = (l < 16) ? 0 : 8;
  int off = (j < 4) ? (ks + 2 * j) : (ks + 16 + 2 * (j - 4));
  int k0 = kc * 32 + off;
  comm[idx] = pack2(h0[(size_t)m * CH + k0], h0[(size_t)m * CH + k0 + 1]);
}

// ---------------------------------------------------------------------------
// Phase 1: xp = X @ Wx + b  (32768x512 * 512x1024), bf16 WMMA, f32 accum.
// 64x64 tile/WG, 8 waves, wave computes 16x32 (two 16x16 accumulators).
// Double-buffered LDS: one barrier per K-chunk, global loads overlap WMMA.
// ---------------------------------------------------------------------------
__global__ __launch_bounds__(256) void xproj_kernel(const float* __restrict__ X,
                                                    const float* __restrict__ Wx,
                                                    const float* __restrict__ bias,
                                                    float* __restrict__ xp) {
  __shared__ unsigned short As[2][64][36];   // [buf][row m][k]  bf16
  __shared__ unsigned short Bs[2][64][36];   // [buf][col n][k]  bf16 (Wx^T tile)

  const int tid  = threadIdx.x;
  const int lane = tid & 31;
  const int wave = tid >> 5;
  const int r0 = blockIdx.y * 64;                 // output row base (t*B+b)
  const int c0 = blockIdx.x * 64;                 // output col base
  const int m0 = (wave >> 1) * 16;                // wave row tile within WG
  const int n0 = (wave & 1) * 32;                 // wave col tile within WG

  // staging registers for global->LDS
  const int acol = tid & 31, arow = tid >> 5;     // A: 64x32 tile
  const int bn   = tid & 63, bkr  = tid >> 6;     // B: 32x64 tile
  float ax[8], bx[8];

  // frag indices
  const int am  = m0 + (lane & 15);
  const int aks = (lane < 16) ? 0 : 8;
  const int fn  = n0 + (lane & 15);
  const int bks = (lane < 16) ? 0 : 8;

  v8f acc0 = {};
  v8f acc1 = {};

  // preload chunk 0
#pragma unroll
  for (int i = 0; i < 8; ++i) {
    ax[i] = X[(size_t)(r0 + arow + 8 * i) * CIN + acol];
    bx[i] = Wx[(size_t)(bkr + 4 * i) * CH + c0 + bn];
  }

  const int NIT = CIN / 32;                        // 16 chunks
  for (int it = 0; it < NIT; ++it) {
    const int p = it & 1;
#pragma unroll
    for (int i = 0; i < 8; ++i) {
      As[p][arow + 8 * i][acol] = f2bf(ax[i]);
      Bs[p][bn][bkr + 4 * i]    = f2bf(bx[i]);
    }
    __syncthreads();
    if (it + 1 < NIT) {
      const int kk = (it + 1) * 32;
#pragma unroll
      for (int i = 0; i < 8; ++i) {
        ax[i] = X[(size_t)(r0 + arow + 8 * i) * CIN + kk + acol];
        bx[i] = Wx[(size_t)(kk + bkr + 4 * i) * CH + c0 + bn];
      }
    }
    Frag a, b0, b1;
#pragma unroll
    for (int j = 0; j < 4; ++j) {
      a.u[j]     = *(const unsigned*)&As[p][am][aks + 2 * j];
      a.u[j + 4] = *(const unsigned*)&As[p][am][aks + 16 + 2 * j];
    }
#pragma unroll
    for (int j = 0; j < 8; ++j) {
      b0.u[j] = *(const unsigned*)&Bs[p][fn][bks + 2 * j + ((j >= 4) ? 8 : 0)];
      b1.u[j] = *(const unsigned*)&Bs[p][fn + 16][bks + 2 * j + ((j >= 4) ? 8 : 0)];
    }
    acc0 = __builtin_amdgcn_wmma_f32_16x16x32_bf16(false, a.v, false, b0.v, (short)0, acc0, false, false);
    acc1 = __builtin_amdgcn_wmma_f32_16x16x32_bf16(false, a.v, false, b1.v, (short)0, acc1, false, false);
    // no second barrier: next iter writes the other buffer; safe by program order
  }

  const int mrow = m0 + ((lane >> 4) << 3);
  const int ncol = c0 + n0 + (lane & 15);
#pragma unroll
  for (int v = 0; v < 8; ++v) {
    int row = r0 + mrow + v;
    xp[(size_t)row * CH + ncol]      = acc0[v] + bias[ncol];
    xp[(size_t)row * CH + ncol + 16] = acc1[v] + bias[ncol + 16];
  }
}

// ---------------------------------------------------------------------------
// Phase 2: sequential scan. Grid (16,4): x = 64-col slice, y = 16-row batch
// group. Groups fully independent; the 16 WGs of a group share h via a
// double-buffered A-fragment buffer in d_ws + L2 atomic barrier.
// Per step, the group's 32KB A-block is staged once into LDS (async copy on
// CDNA5), deduplicating the 4 waves' reads; B frags stream from L2 (hot).
// ---------------------------------------------------------------------------
__global__ __launch_bounds__(128) void rnn_scan_kernel(float* __restrict__ out,
                                                       float* __restrict__ xp,
                                                       const unsigned* __restrict__ whSwz,
                                                       unsigned* __restrict__ comm,
                                                       unsigned* __restrict__ cnt) {
  __shared__ unsigned aFrag[32 * 32 * 8];        // 32KB: [(kc*32+lane)*8+j]
  __shared__ unsigned short hl[16][68];          // bf16 bounce of h_new tile

  const int tid  = threadIdx.x;
  const int lane = tid & 31;
  const int wave = tid >> 5;
  const int c = blockIdx.x;               // column slice (64 cols)
  const int g = blockIdx.y;               // batch group (16 rows)
  const int nt   = c * 4 + wave;          // global 16-col tile id
  const int mlo  = (lane >> 4) << 3;
  const int nloc = lane & 15;
  const int colg = c * 64 + wave * 16 + nloc;

  for (int t = 0; t < T_STEPS; ++t) {
    const unsigned* commR = comm + (size_t)(t & 1) * 32768;
    unsigned*       commW = comm + (size_t)((t & 1) ^ 1) * 32768;
    const unsigned* src   = commR + (size_t)g * 8192;   // this group's A block

    // ---- stage A block (32KB) into LDS, once per WG ----
#if HAVE_ASYNC_LDS
#pragma unroll
    for (int i = 0; i < 16; ++i) {
      int off = (i * 128 + tid) * 4;               // u32 index, 16B granularity
      __builtin_amdgcn_global_load_async_to_lds_b128(
          (glb_v4i_p)(src + off), (lds_v4i_p)(&aFrag[off]), 0, 0);
    }
#if __has_builtin(__builtin_amdgcn_s_wait_asynccnt)
    __builtin_amdgcn_s_wait_asynccnt(0);
#else
    asm volatile("s_wait_asynccnt 0x0" ::: "memory");
#endif
#else
#pragma unroll
    for (int i = 0; i < 16; ++i) {
      int off = (i * 128 + tid) * 4;
      *(uint4*)&aFrag[off] = *(const uint4*)(src + off);
    }
#endif
    __syncthreads();

    // ---- h @ Wh over K = 1024 in 32-wide bf16 WMMA chunks ----
    v8f acc = {};
#pragma unroll 2
    for (int kc = 0; kc < 32; ++kc) {
      Frag a, b;
      const unsigned* ap = &aFrag[((kc * 32 + lane) * 8)];
      const unsigned* bp = whSwz + ((size_t)(nt * 32 + kc) * 32 + lane) * 8;
#pragma unroll
      for (int j = 0; j < 8; ++j) { a.u[j] = ap[j]; b.u[j] = bp[j]; }
      acc = __builtin_amdgcn_wmma_f32_16x16x32_bf16(false, a.v, false, b.v, (short)0, acc, false, false);
    }

    // ---- epilogue: tanh(xp + acc), write both output copies, LDS bounce ----
#pragma unroll
    for (int v = 0; v < 8; ++v) {
      int m   = v + mlo;
      int row = g * 16 + m;
      size_t idx = ((size_t)t * BSZ + row) * CH + colg;
      float s = tanhf(xp[idx] + acc[v]);
      out[idx] = s;          // first output copy
      xp[idx]  = s;          // second copy (xp[t] already consumed by this WG)
      hl[m][wave * 16 + nloc] = f2bf(s);
    }
    __syncthreads();

    // ---- swizzle h_new into A-fragment layout for next step (kc=2c,2c+1) ----
#pragma unroll
    for (int r = 0; r < 4; ++r) {
      int u   = tid * 4 + r;                // [0,512) = 2 kc * 32 lanes * 8 j
      int j   = u & 7;
      int l2  = (u >> 3) & 31;
      int kci = (u >> 8) & 1;
      int m2  = l2 & 15;
      int ks  = (l2 < 16) ? 0 : 8;
      int off = (j < 4) ? (ks + 2 * j) : (ks + 16 + 2 * (j - 4));
      int kl  = kci * 32 + off;
      unsigned val = (unsigned)hl[m2][kl] | ((unsigned)hl[m2][kl + 1] << 16);
      commW[((size_t)(g * 32 + (2 * c + kci)) * 32 + l2) * 8 + j] = val;
    }

    // ---- intra-group barrier (16 WGs) via monotonic L2 counter ----
    __threadfence();          // release: make comm writes device-visible
    __syncthreads();
    if (tid == 0) {
      __hip_atomic_fetch_add(&cnt[g], 1u, __ATOMIC_RELEASE, __HIP_MEMORY_SCOPE_AGENT);
      unsigned target = 16u * (unsigned)(t + 1);
      while (__hip_atomic_load(&cnt[g], __ATOMIC_ACQUIRE, __HIP_MEMORY_SCOPE_AGENT) < target) {
        __builtin_amdgcn_s_sleep(2);
      }
    }
    __syncthreads();
    __threadfence();          // acquire: drop stale L0 lines before next reads
  }
}

// ---------------------------------------------------------------------------
// Launch
// ---------------------------------------------------------------------------
extern "C" void kernel_launch(void* const* d_in, const int* in_sizes, int n_in,
                              void* d_out, int out_size, void* d_ws, size_t ws_size,
                              hipStream_t stream) {
  const float* X    = (const float*)d_in[0];   // [T,B,C_IN]
  const float* h0   = (const float*)d_in[1];   // [1,B,C_H]
  const float* Wx   = (const float*)d_in[2];   // [C_IN,C_H]
  const float* Wh   = (const float*)d_in[3];   // [C_H,C_H]
  const float* bias = (const float*)d_in[4];   // [C_H]

  float* out = (float*)d_out;
  const size_t HALF = (size_t)T_STEPS * BSZ * CH;
  float* xp = out + HALF;                      // stage xp in second output half

  unsigned char* ws = (unsigned char*)d_ws;
  unsigned* cnt   = (unsigned*)(ws);                   // 4 counters (1 KB region)
  unsigned* comm  = (unsigned*)(ws + 1024);            // 2 x 32768 u32 = 256 KB
  unsigned* whSwz = (unsigned*)(ws + 1024 + 262144);   // 524288 u32 = 2 MB

  (void)hipMemsetAsync(cnt, 0, 1024, stream);
  prep_wh_kernel<<<2048, 256, 0, stream>>>(Wh, whSwz);
  prep_h0_kernel<<<128, 256, 0, stream>>>(h0, comm);

  dim3 g1(CH / 64, (T_STEPS * BSZ) / 64);              // 16 x 512 tiles
  xproj_kernel<<<g1, 256, 0, stream>>>(X, Wx, bias, xp);

  rnn_scan_kernel<<<dim3(16, 4), 128, 0, stream>>>(out, xp, whSwz, comm, cnt);
}
